// HGT_73564199846403
// MI455X (gfx1250) — compile-verified, matrix operand order checked by reading, not run on hardware
//
#include <hip/hip_runtime.h>
#include <hip/hip_bf16.h>
#include <stdint.h>

// ---------------------------------------------------------------------------
// HGT (heterogeneous graph transformer) for gfx1250 / MI455X.
// Dense projections run on v_wmma_f32_16x16x32_bf16 (bf16 inputs, f32 accum).
// Edge phase (segment softmax + scatter) is HBM/atomic bound -> VALU kernels.
// B-tiles stream global->LDS via the CDNA5 async-copy path (ASYNCcnt).
// ---------------------------------------------------------------------------

typedef __attribute__((ext_vector_type(16))) __bf16 v16bf;
typedef __attribute__((ext_vector_type(8)))  __bf16 v8bf;
typedef __attribute__((ext_vector_type(8)))  float  v8f;
typedef __attribute__((ext_vector_type(4)))  int    i32x4;

#define AS_GLOBAL __attribute__((address_space(1)))
#define AS_LOCAL  __attribute__((address_space(3)))

#define HIDDEN 256
#define HEADS  8
#define DHEAD  32

#define EPI_NONE 0
#define EPI_RELU 1
#define EPI_SKIP 2

#if __has_builtin(__builtin_amdgcn_global_load_async_to_lds_b128)
#define HAVE_ASYNC_LDS 1
#else
#define HAVE_ASYNC_LDS 0
#endif

__device__ __forceinline__ int   f2ord(float f){ int i=__float_as_int(f); return (i>=0)? i : (i ^ 0x7fffffff); }
__device__ __forceinline__ float ord2f(int i){ return __int_as_float((i>=0)? i : (i ^ 0x7fffffff)); }
__device__ __forceinline__ float gelu_exact(float x){ return 0.5f*x*(1.0f + erff(x*0.7071067811865475f)); }

__device__ __forceinline__ v16bf cat16(v8bf lo, v8bf hi){
  return __builtin_shufflevector(lo, hi, 0,1,2,3,4,5,6,7,8,9,10,11,12,13,14,15);
}

__device__ __forceinline__ void async_wait_all(){
#if HAVE_ASYNC_LDS
#if __has_builtin(__builtin_amdgcn_s_wait_asynccnt)
  __builtin_amdgcn_s_wait_asynccnt(0);
#else
  asm volatile("s_wait_asynccnt 0" ::: "memory");
#endif
#endif
}

#if HAVE_ASYNC_LDS
__device__ __forceinline__ void async_copy_b128(const void* gsrc, void* ldst){
  AS_GLOBAL void* g = (AS_GLOBAL void*)(void*)gsrc;
  AS_LOCAL  void* l = (AS_LOCAL  void*)ldst;
  __builtin_amdgcn_global_load_async_to_lds_b128(
      (AS_GLOBAL i32x4*)g, (AS_LOCAL i32x4*)l, 0, 0);
}
#endif

// ---------------------------------------------------------------------------
// Weight transpose + cast: W[b][K][N] f32  ->  Wt[b][N][K] bf16
// ---------------------------------------------------------------------------
__global__ void transpose_bf16_kernel(const float* __restrict__ W, __bf16* __restrict__ Wt,
                                      int K, int N, long total)
{
  long idx = (long)blockIdx.x * blockDim.x + threadIdx.x;
  if (idx >= total) return;
  long kn = (long)K * N;
  long b  = idx / kn;
  long r  = idx - b * kn;
  int  k  = (int)(r / N);
  int  n  = (int)(r - (long)k * N);
  Wt[b * kn + (long)n * K + k] = (__bf16)W[idx];
}

__global__ void fill_f32_kernel(float* __restrict__ p, float v, long n)
{
  long i = (long)blockIdx.x * blockDim.x + threadIdx.x;
  if (i < n) p[i] = v;
}

__global__ void fill_i32_kernel(int* __restrict__ p, int v, long n)
{
  long i = (long)blockIdx.x * blockDim.x + threadIdx.x;
  if (i < n) p[i] = v;
}

// In-place exact GELU (keeps the transcendental out of the WMMA inner loop).
__global__ void gelu_kernel(float* __restrict__ p, long n)
{
  long i = (long)blockIdx.x * blockDim.x + threadIdx.x;
  if (i < n) p[i] = gelu_exact(p[i]);
}

// ---------------------------------------------------------------------------
// C[M,N] = A[M,K] @ Wt[N,K]^T + bias, fused epilogue.
//   epiMode: EPI_NONE / EPI_RELU / EPI_SKIP (out = s*y + (1-s)*xin, s=sigmoid)
// Block 256 threads = 8 wave32 waves, block tile 128x64, wave tile 32x32
// (2x2 WMMA of 16x16), K stepped by 32 (one bf16 WMMA per step).
// A staged f32->bf16 through VGPRs (float4 loads); B staged with the async
// global->LDS copy path (ASYNCcnt) when the toolchain exposes it.
// ---------------------------------------------------------------------------
__global__ __launch_bounds__(256)
void gemm_bf16_kernel(const float* __restrict__ A, long M, int K,
                      const __bf16* __restrict__ Wt, const float* __restrict__ bias, int N,
                      float* __restrict__ C, int epiMode,
                      const float* __restrict__ xin, const float* __restrict__ skipp)
{
  const int LDT = 40;                       // halves; row pitch 80B (16B aligned)
  __shared__ __bf16 As[128 * LDT];
  __shared__ __bf16 Bs[64  * LDT];

  const int tid   = threadIdx.x;
  const int wid   = tid >> 5;
  const int lane  = tid & 31;
  const int waveM = wid & 3;                // 4 waves along M
  const int waveN = wid >> 2;               // 2 waves along N
  const int m0    = waveM * 32;
  const int n0    = waveN * 32;
  const long blockM = (long)blockIdx.x * 128;
  const long blockN = (long)blockIdx.y * 64;

  const int arow = tid >> 1;                // 0..127
  const int acol = (tid & 1) * 16;
  const int brow = tid >> 2;                // 0..63
  const int bcol = (tid & 3) * 8;

  const bool aInM = (blockM + arow) < M;
  const float* aSrc = A + (blockM + arow) * (long)K + acol;   // + k0 each iter
  const __bf16* bSrc = Wt + (blockN + brow) * (long)K + bcol; // + k0 each iter
  __bf16* aDst = &As[arow * LDT + acol];
  __bf16* bDst = &Bs[brow * LDT + bcol];

  v8f acc[2][2] = {};

  for (int k0 = 0; k0 < K; k0 += 32) {
    // --- stage B tile: bf16 global -> LDS, 16B per lane ---
#if HAVE_ASYNC_LDS
    async_copy_b128(bSrc + k0, bDst);
#else
    *(v8bf*)bDst = *(const v8bf*)(bSrc + k0);
#endif
    // --- stage A tile: f32 -> bf16 through VGPRs (4x global_load_b128) ---
    {
      v8bf lo, hi;
      if (aInM) {
        const float4* s4 = (const float4*)(aSrc + k0);
        if (k0 + 32 < K) __builtin_prefetch(aSrc + k0 + 32, 0, 1);
        float4 f0 = s4[0], f1 = s4[1], f2 = s4[2], f3 = s4[3];
        lo[0]=(__bf16)f0.x; lo[1]=(__bf16)f0.y; lo[2]=(__bf16)f0.z; lo[3]=(__bf16)f0.w;
        lo[4]=(__bf16)f1.x; lo[5]=(__bf16)f1.y; lo[6]=(__bf16)f1.z; lo[7]=(__bf16)f1.w;
        hi[0]=(__bf16)f2.x; hi[1]=(__bf16)f2.y; hi[2]=(__bf16)f2.z; hi[3]=(__bf16)f2.w;
        hi[4]=(__bf16)f3.x; hi[5]=(__bf16)f3.y; hi[6]=(__bf16)f3.z; hi[7]=(__bf16)f3.w;
      } else {
        lo = (v8bf){}; hi = (v8bf){};
      }
      *(v8bf*)aDst       = lo;
      *(v8bf*)(aDst + 8) = hi;
    }
    async_wait_all();
    __syncthreads();

    // --- fragments per ISA 7.12.2 (16-bit A 16x32 layout) ---
    const int half = lane >> 4;
    const int l15  = lane & 15;
    v16bf afrag[2], bfrag[2];
    #pragma unroll
    for (int i = 0; i < 2; i++) {
      const __bf16* p = &As[(m0 + 16*i + l15) * LDT];
      v8bf lo = *(const v8bf*)(p + half * 8);
      v8bf hi = *(const v8bf*)(p + 16 + half * 8);
      afrag[i] = cat16(lo, hi);
    }
    #pragma unroll
    for (int j = 0; j < 2; j++) {
      const __bf16* p = &Bs[(n0 + 16*j + l15) * LDT];
      v8bf lo = *(const v8bf*)(p + half * 8);
      v8bf hi = *(const v8bf*)(p + 16 + half * 8);
      bfrag[j] = cat16(lo, hi);
    }
    #pragma unroll
    for (int i = 0; i < 2; i++)
      #pragma unroll
      for (int j = 0; j < 2; j++)
        acc[i][j] = __builtin_amdgcn_wmma_f32_16x16x32_bf16(
            false, afrag[i], false, bfrag[j], (short)0, acc[i][j], false, false);
    __syncthreads();
  }

  // --- epilogue: C layout = M:(vgpr + 8*(lane>=16)), N:(lane&15) ---
  float sblend = 0.f;
  if (epiMode == EPI_SKIP) { float sv = skipp[0]; sblend = 1.f / (1.f + __expf(-sv)); }
  const int half = lane >> 4;
  const int l15  = lane & 15;
  #pragma unroll
  for (int i = 0; i < 2; i++) {
    #pragma unroll
    for (int j = 0; j < 2; j++) {
      long col   = blockN + n0 + 16*j + l15;
      float bv   = bias[col];
      long mBase = blockM + m0 + 16*i + half * 8;
      #pragma unroll
      for (int r = 0; r < 8; r++) {
        long row = mBase + r;
        if (row < M) {
          float v = acc[i][j][r] + bv;
          if (epiMode == EPI_RELU)      v = fmaxf(v, 0.f);
          else if (epiMode == EPI_SKIP) v = sblend * v + (1.f - sblend) * xin[row * (long)N + col];
          C[row * (long)N + col] = v;
        }
      }
    }
  }
}

// ---------------------------------------------------------------------------
// Relation transform: Y[n, 32h+e] = sum_d X[n, 32h+d] * At[h][e][d]
// One head per blockIdx.y; block of 8 waves handles 128 rows (16 rows/wave).
// K=32 -> exactly one bf16 WMMA per 16x16 output tile (2 tiles per wave).
// ---------------------------------------------------------------------------
__global__ __launch_bounds__(256)
void rel_gemm_kernel(const float* __restrict__ X, const __bf16* __restrict__ At,
                     float* __restrict__ Y, long Nn)
{
  const int wid  = threadIdx.x >> 5;
  const int lane = threadIdx.x & 31;
  const int h    = blockIdx.y;
  const long rb  = (long)blockIdx.x * 128 + wid * 16;
  if (rb >= Nn) return;                      // wave-uniform: EXEC stays full
  const int half = lane >> 4;
  const int l15  = lane & 15;

  long row = rb + l15; if (row >= Nn) row = Nn - 1;
  const float4* base4 = (const float4*)(X + row * (long)HIDDEN + h * DHEAD);
  float4 f0 = base4[half * 2], f1 = base4[half * 2 + 1];
  float4 f2 = base4[4 + half * 2], f3 = base4[4 + half * 2 + 1];
  v16bf a;
  a[0]=(__bf16)f0.x;  a[1]=(__bf16)f0.y;  a[2]=(__bf16)f0.z;  a[3]=(__bf16)f0.w;
  a[4]=(__bf16)f1.x;  a[5]=(__bf16)f1.y;  a[6]=(__bf16)f1.z;  a[7]=(__bf16)f1.w;
  a[8]=(__bf16)f2.x;  a[9]=(__bf16)f2.y;  a[10]=(__bf16)f2.z; a[11]=(__bf16)f2.w;
  a[12]=(__bf16)f3.x; a[13]=(__bf16)f3.y; a[14]=(__bf16)f3.z; a[15]=(__bf16)f3.w;

  #pragma unroll
  for (int j = 0; j < 2; j++) {
    const __bf16* bp = At + (long)h * (DHEAD * DHEAD) + (16*j + l15) * DHEAD;
    v8bf lo = *(const v8bf*)(bp + half * 8);
    v8bf hi = *(const v8bf*)(bp + 16 + half * 8);
    v16bf b = cat16(lo, hi);
    v8f c = {};
    c = __builtin_amdgcn_wmma_f32_16x16x32_bf16(false, a, false, b, (short)0, c, false, false);
    long mBase = rb + half * 8;
    int  col   = h * DHEAD + 16*j + l15;
    #pragma unroll
    for (int r = 0; r < 8; r++) {
      long orow = mBase + r;
      if (orow < Nn) Y[orow * (long)HIDDEN + col] = c[r];
    }
  }
}

// ---------------------------------------------------------------------------
// Edge phase: scores -> segment max (int-punned atomicMax) -> exp/segment sum
// -> normalized scatter-add of messages. Wave-per-edge layouts.
// ---------------------------------------------------------------------------
__global__ __launch_bounds__(256)
void scores_kernel(const int* __restrict__ src, const int* __restrict__ dst,
                   const float* __restrict__ Q, const float* __restrict__ KR,
                   const float* __restrict__ prio, float* __restrict__ sc,
                   int* __restrict__ smax, int E)
{
  int e = blockIdx.x * 8 + (threadIdx.x >> 5);
  if (e >= E) return;
  int lane = threadIdx.x & 31;
  long s = src[e], d = dst[e];
  float part[HEADS];
  #pragma unroll
  for (int j = 0; j < HEADS; j++)
    part[j] = Q[d * HIDDEN + j * DHEAD + lane] * KR[s * HIDDEN + j * DHEAD + lane];
  #pragma unroll
  for (int off = 16; off > 0; off >>= 1)
    #pragma unroll
    for (int j = 0; j < HEADS; j++) part[j] += __shfl_xor(part[j], off, 32);
  if (lane < HEADS) {
    float v = part[lane] * prio[lane] * 0.1767766952966369f;  // 1/sqrt(32)
    sc[(long)e * HEADS + lane] = v;
    atomicMax(&smax[d * HEADS + lane], f2ord(v));
  }
}

__global__ __launch_bounds__(256)
void expsum_kernel(const int* __restrict__ dst, float* __restrict__ sc,
                   const int* __restrict__ smax, float* __restrict__ ssum, int E)
{
  long idx = (long)blockIdx.x * blockDim.x + threadIdx.x;
  if (idx >= (long)E * HEADS) return;
  int e = (int)(idx >> 3), h = (int)(idx & 7);
  long d = dst[e];
  float m  = ord2f(smax[d * HEADS + h]);
  float ex = __expf(sc[idx] - m);
  sc[idx] = ex;
  atomicAdd(&ssum[d * HEADS + h], ex);
}

__global__ __launch_bounds__(256)
void scatter_kernel(const int* __restrict__ src, const int* __restrict__ dst,
                    const float* __restrict__ VR, const float* __restrict__ sc,
                    const float* __restrict__ ssum, float* __restrict__ agg, int E)
{
  int e = blockIdx.x * 8 + (threadIdx.x >> 5);
  if (e >= E) return;
  int lane = threadIdx.x & 31;
  long s = src[e], d = dst[e];
  #pragma unroll
  for (int j = 0; j < HEADS; j++) {
    float alpha = sc[(long)e * HEADS + j] / (ssum[d * HEADS + j] + 1e-16f);
    atomicAdd(&agg[d * HIDDEN + j * DHEAD + lane],
              VR[s * HIDDEN + j * DHEAD + lane] * alpha);
  }
}

// ---------------------------------------------------------------------------
// Host orchestration. Param leaf layout assumes jax pytree flattening with
// sorted dict keys:
//  d_in[0..2]  x_text, x_emotion, x_intensity
//  d_in[3..8]  in: emotion{W,b}, intensity{W,b}, text{W,b}
//  per layer (39 leaves, layer l at base = 9 + 39*l):
//    +0  a:   emo{W,b}, int{W,b}, text{W,b}
//    +6  k:   ...    +12 q: ...
//    +18 rel: e2t{a,m,p}, i2t{a,m,p}, t2e{a,m,p}, t2i{a,m,p}
//    +30 skip: emo, int, text     +33 v: ...
//  last 8: src/dst for t2e, e2t, t2i, i2t
// ---------------------------------------------------------------------------
extern "C" void kernel_launch(void* const* d_in, const int* in_sizes, int n_in,
                              void* d_out, int out_size, void* d_ws, size_t ws_size,
                              hipStream_t stream)
{
  (void)out_size; (void)ws_size;
  const float* xraw[3] = {(const float*)d_in[0], (const float*)d_in[1], (const float*)d_in[2]};
  const int Kin[3] = {512, 256, 256};
  long Nn[3] = { in_sizes[0] / 512, in_sizes[1] / 256, in_sizes[2] / 256 };
  long off[3] = { 0, Nn[0], Nn[0] + Nn[1] };
  long Ntot = Nn[0] + Nn[1] + Nn[2];
  long maxN = Nn[0]; if (Nn[1] > maxN) maxN = Nn[1]; if (Nn[2] > maxN) maxN = Nn[2];
  int eb = n_in - 8;
  int E  = in_sizes[eb];

  // ---- workspace carve ----
  uintptr_t w = (uintptr_t)d_ws;
  auto carve = [&](size_t bytes) -> void* {
    void* p = (void*)w; w += (bytes + 255) & ~(size_t)255; return p;
  };
  float* X0  = (float*)carve((size_t)Ntot * HIDDEN * 4);
  float* X1  = (float*)carve((size_t)Ntot * HIDDEN * 4);
  float* Kb  = (float*)carve((size_t)Ntot * HIDDEN * 4);
  float* Qb  = (float*)carve((size_t)Ntot * HIDDEN * 4);
  float* Vb  = (float*)carve((size_t)Ntot * HIDDEN * 4);
  float* KR  = (float*)carve((size_t)maxN * HIDDEN * 4);
  float* VR  = (float*)carve((size_t)maxN * HIDDEN * 4);
  float* AGG = (float*)carve((size_t)Ntot * HIDDEN * 4);
  float* SC  = (float*)carve((size_t)E * HEADS * 4);
  int*   SMAX= (int*)  carve((size_t)maxN * HEADS * 4);
  float* SSUM= (float*)carve((size_t)maxN * HEADS * 4);
  __bf16* WT0 = (__bf16*)carve((size_t)512 * 256 * 2);
  __bf16* WT1 = (__bf16*)carve((size_t)HEADS * 32 * 32 * 2);
  __bf16* WT2 = (__bf16*)carve((size_t)HEADS * 32 * 32 * 2);

  auto tr = [&](const void* W, __bf16* Wt, int K, int N, int B) {
    long total = (long)B * K * N;
    transpose_bf16_kernel<<<dim3((unsigned)((total + 255) / 256)), 256, 0, stream>>>(
        (const float*)W, Wt, K, N, total);
  };
  auto gemm = [&](const float* A, long M, int K, const __bf16* Wt, const void* bias,
                  float* C, int epi, const float* xin, const void* skipp) {
    dim3 g((unsigned)((M + 127) / 128), HIDDEN / 64);
    gemm_bf16_kernel<<<g, 256, 0, stream>>>(A, M, K, Wt, (const float*)bias, HIDDEN,
                                            C, epi, xin, (const float*)skipp);
  };

  static const int slot[3] = {2, 0, 1};  // type (text,emo,int) -> sorted pos (emo,int,text)

  // ---- input projections with fused ReLU ----
  for (int t = 0; t < 3; t++) {
    tr(d_in[3 + slot[t] * 2], WT0, Kin[t], HIDDEN, 1);
    gemm(xraw[t], Nn[t], Kin[t], WT0, d_in[3 + slot[t] * 2 + 1],
         X0 + off[t] * HIDDEN, EPI_RELU, nullptr, nullptr);
  }

  // ---- HGT layers ----
  for (int l = 0; l < 2; l++) {
    int base = 9 + l * 39;
    float* Xin  = l ? X1 : X0;
    float* Xout = l ? (float*)d_out : X1;

    for (int t = 0; t < 3; t++) {
      long o = off[t] * HIDDEN;
      tr(d_in[base + 6  + slot[t] * 2], WT0, 256, 256, 1);   // k
      gemm(Xin + o, Nn[t], 256, WT0, d_in[base + 6  + slot[t] * 2 + 1], Kb + o, EPI_NONE, nullptr, nullptr);
      tr(d_in[base + 12 + slot[t] * 2], WT0, 256, 256, 1);   // q
      gemm(Xin + o, Nn[t], 256, WT0, d_in[base + 12 + slot[t] * 2 + 1], Qb + o, EPI_NONE, nullptr, nullptr);
      tr(d_in[base + 33 + slot[t] * 2], WT0, 256, 256, 1);   // v
      gemm(Xin + o, Nn[t], 256, WT0, d_in[base + 33 + slot[t] * 2 + 1], Vb + o, EPI_NONE, nullptr, nullptr);
    }

    fill_f32_kernel<<<(unsigned)((Ntot * HIDDEN + 255) / 256), 256, 0, stream>>>(AGG, 0.f, Ntot * HIDDEN);

    struct RelT { int s, d, rslot, ei; };
    const RelT rels[4] = { {0,1,2,eb+0},   // t2e
                           {1,0,0,eb+2},   // e2t
                           {0,2,3,eb+4},   // t2i
                           {2,0,1,eb+6} }; // i2t
    for (int r = 0; r < 4; r++) {
      const RelT& R = rels[r];
      const int* si = (const int*)d_in[R.ei];
      const int* di = (const int*)d_in[R.ei + 1];
      int rb = base + 18 + R.rslot * 3;
      tr(d_in[rb + 0], WT1, 32, 32, HEADS);   // rel 'a' -> [h][e][d] bf16
      rel_gemm_kernel<<<dim3((unsigned)((Nn[R.s] + 127) / 128), HEADS), 256, 0, stream>>>(
          Kb + off[R.s] * HIDDEN, WT1, KR, Nn[R.s]);
      tr(d_in[rb + 1], WT2, 32, 32, HEADS);   // rel 'm'
      rel_gemm_kernel<<<dim3((unsigned)((Nn[R.s] + 127) / 128), HEADS), 256, 0, stream>>>(
          Vb + off[R.s] * HIDDEN, WT2, VR, Nn[R.s]);

      fill_i32_kernel<<<(unsigned)((Nn[R.d] * HEADS + 255) / 256), 256, 0, stream>>>(
          SMAX, (int)0x80000000, Nn[R.d] * HEADS);
      fill_f32_kernel<<<(unsigned)((Nn[R.d] * HEADS + 255) / 256), 256, 0, stream>>>(
          SSUM, 0.f, Nn[R.d] * HEADS);

      scores_kernel<<<(unsigned)((E + 7) / 8), 256, 0, stream>>>(
          si, di, Qb + off[R.d] * HIDDEN, KR, (const float*)d_in[rb + 2], SC, SMAX, E);
      expsum_kernel<<<(unsigned)(((long)E * HEADS + 255) / 256), 256, 0, stream>>>(
          di, SC, SMAX, SSUM, E);
      scatter_kernel<<<(unsigned)((E + 7) / 8), 256, 0, stream>>>(
          si, di, VR, SC, SSUM, AGG + off[R.d] * HIDDEN, E);
    }

    // out projection: GELU(agg) @ Wa + b, sigmoid-skip blend with Xin.
    // GELU done as a separate cheap elementwise pass (keeps erff out of the
    // WMMA K-loop).
    gelu_kernel<<<(unsigned)((Ntot * HIDDEN + 255) / 256), 256, 0, stream>>>(AGG, Ntot * HIDDEN);
    for (int t = 0; t < 3; t++) {
      long o = off[t] * HIDDEN;
      tr(d_in[base + slot[t] * 2], WT0, 256, 256, 1);
      gemm(AGG + o, Nn[t], 256, WT0, d_in[base + slot[t] * 2 + 1],
           Xout + o, EPI_SKIP, Xin + o, d_in[base + 30 + slot[t]]);
    }
  }
}
